// StigmergyEngine_37752762532579
// MI455X (gfx1250) — matrix operand order repaired and checked by reading, not compiled, not run on hardware
//
#include <hip/hip_runtime.h>
#include <hip/hip_bf16.h>
#include <math.h>

// ---------------------------------------------------------------------------
// Problem constants
// ---------------------------------------------------------------------------
constexpr int N_AG = 65536, GS = 512, NP_ = 4;
constexpr int IN_ = 256, HID_ = 512, OUT_ = 256, H1_ = 128, G3_ = 1536;
#define PI_F 3.14159265358979323846f

// ---------------------------------------------------------------------------
// WMMA types / helpers (CDNA5 gfx1250, wave32)
// ---------------------------------------------------------------------------
typedef __attribute__((ext_vector_type(16))) _Float16 v16h;
typedef __attribute__((ext_vector_type(8)))  _Float16 v8h;
typedef __attribute__((ext_vector_type(8)))  float    v8f;

__device__ inline v8f wmma16(v16h a, v16h b, v8f c) {
  return __builtin_amdgcn_wmma_f32_16x16x32_f16(
      /*neg_a=*/false, a, /*neg_b=*/false, b,
      /*c_mod=*/(short)0, c, /*reuse_a=*/false, /*reuse_b=*/false);
}

// A-matrix fragment (16x32 f16): lane holds row (lane&15); K = {hi..hi+7, 16+hi..16+hi+7},
// hi = 8*(lane>=16).  Two 16-byte LDS reads per fragment.
__device__ inline v16h ldfragA(const _Float16* base, int rowoff, int kbase,
                               int stride, int lane) {
  const _Float16* p = base + (size_t)(rowoff + (lane & 15)) * stride
                           + kbase + ((lane >> 4) << 3);
  v16h a;
  *(v8h*)&a       = *(const v8h*)p;
  *((v8h*)&a + 1) = *(const v8h*)(p + 16);
  return a;
}

// B-matrix fragment from repacked weights: 32 lanes x 16 f16 contiguous (1KB/frag)
__device__ inline v16h ldfragB(const _Float16* wp, size_t frag, int lane) {
  return *(const v16h*)(wp + frag * 512 + (size_t)lane * 16);
}

// ---------------------------------------------------------------------------
// Workspace layout (bytes)
// ---------------------------------------------------------------------------
constexpr size_t FRAG_B  = 1024;
constexpr size_t SZ_WA1P = (768 / 32) * (128 / 16) * FRAG_B;   // 196608
constexpr size_t SZ_WA2P = (128 / 32) * (256 / 16) * FRAG_B;   // 65536
constexpr size_t SZ_WIHP = (256 / 32) * (1536 / 16) * FRAG_B;  // 786432
constexpr size_t SZ_WHHP = (512 / 32) * (1536 / 16) * FRAG_B;  // 1572864

constexpr size_t O_WA1P  = 0;
constexpr size_t O_WG1P  = O_WA1P + SZ_WA1P;
constexpr size_t O_WA2P  = O_WG1P + SZ_WA1P;
constexpr size_t O_WG2P  = O_WA2P + SZ_WA2P;
constexpr size_t O_WIHP  = O_WG2P + SZ_WA2P;
constexpr size_t O_WHHP  = O_WIHP + SZ_WIHP;
constexpr size_t O_TENS  = O_WHHP + SZ_WHHP;                 // N floats
constexpr size_t O_HEAD  = O_TENS + (size_t)N_AG * 4;        // N floats
constexpr size_t O_NPOS  = O_HEAD + (size_t)N_AG * 4;        // N int2
constexpr size_t O_GRIDW = O_NPOS + (size_t)N_AG * 8;        // 4*512*512 floats
constexpr size_t O_QMEAN = O_GRIDW + (size_t)NP_ * GS * GS * 4;  // 16*512
constexpr size_t O_GO    = O_QMEAN + 16 * 512 * 4;               // 512
// zero-initialized accumulator region (contiguous):
constexpr size_t O_ZERO  = O_GO + 512 * 4;
constexpr size_t O_QSUM  = O_ZERO;                 // 16*512 floats
constexpr size_t O_QCNT  = O_QSUM + 16 * 512 * 4;  // 64 floats (16 used)
constexpr size_t O_FAM   = O_QCNT + 64 * 4;        // 8*512 floats
constexpr size_t O_RED   = O_FAM + 8 * 512 * 4;    // 64 floats
constexpr size_t O_COMB  = O_RED + 64 * 4;         // 256 floats
constexpr size_t ZERO_FLOATS = (O_COMB + 256 * 4 - O_ZERO) / 4;
constexpr size_t O_OUTWS = O_COMB + 256 * 4;       // N*256 floats (out matrix)
// red[] indices: 0 sinsum, 1 cossum, 2 dsum, 3 mh, 4 tmax(bits), 5 sumexp, 6 tsum

// ---------------------------------------------------------------------------
__global__ void k_zero(float* p, int n) {
  int i = blockIdx.x * 256 + threadIdx.x;
  if (i < n) p[i] = 0.f;
}

// Repack f32 weight (O x rowstride, using first ktiles*32 cols) into per-lane
// f16 B fragments: frag f = no*ktiles + ko, lane l: W[no*16+(l&15)][ko*32+(l>>4)*8 + (j<8 ? j : j+8)]
__global__ void k_repack(const float* __restrict__ W, _Float16* __restrict__ outp,
                         int ntiles, int ktiles, int rowstride) {
  int idx = blockIdx.x * blockDim.x + threadIdx.x;
  int total = ntiles * ktiles * 32;
  if (idx >= total) return;
  int lane = idx & 31, f = idx >> 5;
  int ko = f % ktiles, no = f / ktiles;
  const float* src = W + (size_t)(no * 16 + (lane & 15)) * rowstride
                       + ko * 32 + ((lane >> 4) << 3);
  _Float16* dst = outp + (size_t)f * 512 + lane * 16;
#pragma unroll
  for (int j = 0; j < 16; ++j) dst[j] = (_Float16)src[j < 8 ? j : j + 8];
}

// ---------------------------------------------------------------------------
__device__ inline float block_reduce_sum(float v, float* sb) {
  int tid = threadIdx.x;
  sb[tid] = v; __syncthreads();
  for (int s = 128; s > 0; s >>= 1) {
    if (tid < s) sb[tid] += sb[tid + s];
    __syncthreads();
  }
  float r = sb[0]; __syncthreads();
  return r;
}

// ---------------------------------------------------------------------------
// Stigmergy: heading loop (4 pheromone layers), new positions, sin/cos sums
// ---------------------------------------------------------------------------
__global__ __launch_bounds__(256) void k_agents(
    const float* __restrict__ grid, const int* __restrict__ pos,
    const float* __restrict__ headings, const float* __restrict__ noise,
    float* __restrict__ head_out, int* __restrict__ npos, float* red) {
  __shared__ float sb[256];
  int i = blockIdx.x * 256 + threadIdx.x;
  float px = (float)pos[2 * i], py = (float)pos[2 * i + 1];
  float h = headings[i];
  for (int pt = 0; pt < NP_; ++pt) {
    const float* g = grid + (size_t)pt * GS * GS;
    float v[3]; float angs[3] = {h, h + 0.5f, h - 0.5f};
#pragma unroll
    for (int s = 0; s < 3; ++s) {
      float gx = fmodf(px + cosf(angs[s]) * 2.0f, (float)GS);
      if (gx < 0.f) gx += GS; if (gx >= (float)GS) gx = 0.f;
      float gy = fmodf(py + sinf(angs[s]) * 2.0f, (float)GS);
      if (gy < 0.f) gy += GS; if (gy >= (float)GS) gy = 0.f;
      v[s] = g[(int)gx * GS + (int)gy];
    }
    float turn = (v[1] > v[0] && v[1] > v[2]) ? 0.3f
               : ((v[2] > v[0] && v[2] > v[1]) ? -0.3f : 0.f);
    h += turn + 0.1f * noise[(size_t)pt * N_AG + i];
  }
  head_out[i] = h;
  int dx = (int)truncf(cosf(h) * 1.5f);
  int dy = (int)truncf(sinf(h) * 1.5f);
  int nx = ((pos[2 * i] + dx) % GS + GS) % GS;
  int ny = ((pos[2 * i + 1] + dy) % GS + GS) % GS;
  npos[2 * i] = nx; npos[2 * i + 1] = ny;
  float ss = block_reduce_sum(sinf(h), sb);
  float cs = block_reduce_sum(cosf(h), sb);
  if (threadIdx.x == 0) { atomicAdd(&red[0], ss); atomicAdd(&red[1], cs); }
}

__global__ void k_mh(float* red) {
  if (threadIdx.x == 0 && blockIdx.x == 0) red[3] = atan2f(red[0], red[1]);
}

__global__ __launch_bounds__(256) void k_coord(const float* __restrict__ head,
                                               float* red) {
  __shared__ float sb[256];
  int i = blockIdx.x * 256 + threadIdx.x;
  float mh = red[3];
  float m = fmodf(head[i] - mh + PI_F, 2.f * PI_F);
  if (m < 0.f) m += 2.f * PI_F;
  float s = block_reduce_sum(fabsf(m - PI_F), sb);
  if (threadIdx.x == 0) atomicAdd(&red[2], s);
}

// ---------------------------------------------------------------------------
// Grid: copy, deposit, evaporate+diffuse (torus)
// ---------------------------------------------------------------------------
__global__ void k_gridcopy(const float* __restrict__ g, float* __restrict__ gw) {
  int i = blockIdx.x * 256 + threadIdx.x;
  gw[i] = g[i];
}

__global__ void k_deposit(const int* __restrict__ ptype, const int* __restrict__ npos,
                          float* __restrict__ gw) {
  int i = blockIdx.x * 256 + threadIdx.x;
  atomicAdd(&gw[(size_t)ptype[i] * GS * GS + (size_t)npos[2 * i] * GS + npos[2 * i + 1]], 1.0f);
}

__global__ void k_diffuse(const float* __restrict__ gw, float* __restrict__ gout) {
  int idx = blockIdx.x * 256 + threadIdx.x;
  int pt = idx >> 18, rem = idx & (GS * GS - 1);
  int xx = rem >> 9, yy = rem & 511;
  const float* g = gw + (size_t)pt * GS * GS;
  float c  = g[xx * GS + yy];
  float xm = g[((xx + GS - 1) & 511) * GS + yy];
  float xp = g[((xx + 1) & 511) * GS + yy];
  float ym = g[xx * GS + ((yy + GS - 1) & 511)];
  float yp = g[xx * GS + ((yy + 1) & 511)];
  gout[idx] = 0.95f * (0.9f * c + 0.1f * 0.25f * (xm + xp + ym + yp));
}

// ---------------------------------------------------------------------------
// Fused MLP + GRU core: 32 agents / block, 8 waves, all GEMMs via WMMA f16
// ---------------------------------------------------------------------------
__global__ __launch_bounds__(256) void k_core(
    const float* __restrict__ x, const float* __restrict__ hiddens,
    const float* __restrict__ Wih,
    const float* __restrict__ ba1, const float* __restrict__ bg1,
    const float* __restrict__ ba2, const float* __restrict__ bg2,
    const float* __restrict__ bih, const float* __restrict__ bhh,
    const _Float16* __restrict__ wa1p, const _Float16* __restrict__ wg1p,
    const _Float16* __restrict__ wa2p, const _Float16* __restrict__ wg2p,
    const _Float16* __restrict__ wihp, const _Float16* __restrict__ whhp,
    const int* __restrict__ npos, const float* __restrict__ outgrid,
    float* __restrict__ out_ws, float* __restrict__ tens_ws,
    float* __restrict__ newh) {
  __shared__ __align__(16) _Float16 smem[25600];
  __shared__ float tensL[32];
  __shared__ float boostL[32];
  _Float16* xL     = smem;                 // 256  (x as f16, shared rows)
  _Float16* combHL = smem + 256;           // 32 x 520 (hiddens f16)
  _Float16* h1aL   = smem + 16896;         // 32 x 136
  _Float16* h1gL   = h1aL + 32 * 136;      // 32 x 136
  _Float16* outL   = smem + 16896;         // overlay: 32 x 264 (after stage 2)

  const int tid = threadIdx.x, lane = tid & 31, wave = tid >> 5;
  const int row0 = blockIdx.x * 32;

  // ---- stage 0: stage activations, boost factors --------------------------
  for (int c = tid; c < IN_; c += 256) xL[c] = (_Float16)x[c];
  for (int i = tid; i < 32 * HID_; i += 256) {
    int r = i >> 9, c = i & 511;
    combHL[r * 520 + c] = (_Float16)hiddens[(size_t)(row0 + r) * HID_ + c];
  }
  if (tid < 32) {
    tensL[tid] = 0.f;
    int a = row0 + tid;
    int nx = npos[2 * a], ny = npos[2 * a + 1];
    float local = 0.f;
    for (int pt = 0; pt < NP_; ++pt)
      local += outgrid[(size_t)pt * GS * GS + (size_t)nx * GS + ny];
    boostL[tid] = 0.9f + 0.1f * tanhf(local * 0.2f);
  }
  __syncthreads();

  // ---- stage 1: h1 = relu([x|h] @ W1^T + b1), a & g branches --------------
  {
    const int nt = wave;  // one of 8 N-tiles of 128 outputs
    v8f aA0 = {}, aA1 = {}, aG0 = {}, aG1 = {};
    for (int ko = 0; ko < 8; ++ko) {  // K = x part (rows identical)
      v16h a0 = ldfragA(xL, 0, ko * 32, 0, lane);
      v16h bA = ldfragB(wa1p, (size_t)nt * 24 + ko, lane);
      v16h bG = ldfragB(wg1p, (size_t)nt * 24 + ko, lane);
      aA0 = wmma16(a0, bA, aA0); aA1 = wmma16(a0, bA, aA1);
      aG0 = wmma16(a0, bG, aG0); aG1 = wmma16(a0, bG, aG1);
    }
    for (int ko = 0; ko < 16; ++ko) {  // K = hiddens part
      v16h a0 = ldfragA(combHL, 0, ko * 32, 520, lane);
      v16h a1 = ldfragA(combHL, 16, ko * 32, 520, lane);
      v16h bA = ldfragB(wa1p, (size_t)nt * 24 + 8 + ko, lane);
      v16h bG = ldfragB(wg1p, (size_t)nt * 24 + 8 + ko, lane);
      aA0 = wmma16(a0, bA, aA0); aA1 = wmma16(a1, bA, aA1);
      aG0 = wmma16(a0, bG, aG0); aG1 = wmma16(a1, bG, aG1);
    }
    int col = nt * 16 + (lane & 15);
    float bA = ba1[col], bG = bg1[col];
    int mrow = (lane >> 4) * 8;
#pragma unroll
    for (int v = 0; v < 8; ++v) {
      float v0 = aA0[v] + bA, v1 = aA1[v] + bA;
      float w0 = aG0[v] + bG, w1 = aG1[v] + bG;
      h1aL[(mrow + v) * 136 + col]      = (_Float16)(v0 > 0.f ? v0 : 0.f);
      h1aL[(16 + mrow + v) * 136 + col] = (_Float16)(v1 > 0.f ? v1 : 0.f);
      h1gL[(mrow + v) * 136 + col]      = (_Float16)(w0 > 0.f ? w0 : 0.f);
      h1gL[(16 + mrow + v) * 136 + col] = (_Float16)(w1 > 0.f ? w1 : 0.f);
    }
  }
  __syncthreads();

  // ---- stage 2: out = a - g ; tension ------------------------------------
  {
    v8f acc[2][4] = {};
#pragma unroll
    for (int ni = 0; ni < 2; ++ni) {
      const int nt = wave + ni * 8;  // 16 N-tiles of 256 outputs
      for (int ko = 0; ko < 4; ++ko) {
        v16h fa0 = ldfragA(h1aL, 0, ko * 32, 136, lane);
        v16h fa1 = ldfragA(h1aL, 16, ko * 32, 136, lane);
        v16h fg0 = ldfragA(h1gL, 0, ko * 32, 136, lane);
        v16h fg1 = ldfragA(h1gL, 16, ko * 32, 136, lane);
        v16h bA = ldfragB(wa2p, (size_t)nt * 4 + ko, lane);
        v16h bG = ldfragB(wg2p, (size_t)nt * 4 + ko, lane);
        acc[ni][0] = wmma16(fa0, bA, acc[ni][0]);
        acc[ni][1] = wmma16(fa1, bA, acc[ni][1]);
        acc[ni][2] = wmma16(fg0, bG, acc[ni][2]);
        acc[ni][3] = wmma16(fg1, bG, acc[ni][3]);
      }
    }
    __syncthreads();  // all h1 reads done; outL overlays h1 region
    const int mrow = (lane >> 4) * 8;
#pragma unroll
    for (int ni = 0; ni < 2; ++ni) {
      const int nt = wave + ni * 8;
      int col = nt * 16 + (lane & 15);
      float bo = ba2[col] - bg2[col];
#pragma unroll
      for (int v = 0; v < 8; ++v) {
        float o0 = acc[ni][0][v] - acc[ni][2][v] + bo;
        float o1 = acc[ni][1][v] - acc[ni][3][v] + bo;
        outL[(mrow + v) * 264 + col]      = (_Float16)o0;
        outL[(16 + mrow + v) * 264 + col] = (_Float16)o1;
        out_ws[(size_t)(row0 + mrow + v) * OUT_ + col]      = o0;
        out_ws[(size_t)(row0 + 16 + mrow + v) * OUT_ + col] = o1;
        atomicAdd(&tensL[mrow + v], o0 * o0);
        atomicAdd(&tensL[16 + mrow + v], o1 * o1);
      }
    }
  }
  __syncthreads();
  if (tid < 32) {
    float t = tensL[tid] * (1.f / OUT_);
    tensL[tid] = t;
    tens_ws[row0 + tid] = t;
  }
  __syncthreads();

  // ---- stage 3: GRU. Each wave owns 4 j-tiles of 16 hidden cols ----------
  for (int jj = 0; jj < 4; ++jj) {
    const int jt = wave * 4 + jj;  // 0..31; gate tiles jt, jt+32, jt+64
    v8f r0 = {}, r1 = {}, z0 = {}, z1 = {}, i0 = {}, i1 = {}, hn0 = {}, hn1 = {};
    for (int ko = 0; ko < 8; ++ko) {  // gi: K = 256 over out
      v16h a0 = ldfragA(outL, 0, ko * 32, 264, lane);
      v16h a1 = ldfragA(outL, 16, ko * 32, 264, lane);
      v16h br = ldfragB(wihp, (size_t)jt * 8 + ko, lane);
      v16h bz = ldfragB(wihp, (size_t)(jt + 32) * 8 + ko, lane);
      v16h bn = ldfragB(wihp, (size_t)(jt + 64) * 8 + ko, lane);
      if (ko + 1 < 8) __builtin_prefetch(wihp + ((size_t)jt * 8 + ko + 1) * 512 + lane * 16, 0, 3);
      r0 = wmma16(a0, br, r0); r1 = wmma16(a1, br, r1);
      z0 = wmma16(a0, bz, z0); z1 = wmma16(a1, bz, z1);
      i0 = wmma16(a0, bn, i0); i1 = wmma16(a1, bn, i1);
    }
    for (int ko = 0; ko < 16; ++ko) {  // gh: K = 512 over hiddens
      v16h a0 = ldfragA(combHL, 0, ko * 32, 520, lane);
      v16h a1 = ldfragA(combHL, 16, ko * 32, 520, lane);
      v16h br = ldfragB(whhp, (size_t)jt * 16 + ko, lane);
      v16h bz = ldfragB(whhp, (size_t)(jt + 32) * 16 + ko, lane);
      v16h bn = ldfragB(whhp, (size_t)(jt + 64) * 16 + ko, lane);
      if (ko + 1 < 16) __builtin_prefetch(whhp + ((size_t)jt * 16 + ko + 1) * 512 + lane * 16, 0, 3);
      r0 = wmma16(a0, br, r0); r1 = wmma16(a1, br, r1);
      z0 = wmma16(a0, bz, z0); z1 = wmma16(a1, bz, z1);
      hn0 = wmma16(a0, bn, hn0); hn1 = wmma16(a1, bn, hn1);
    }
    const int col = lane & 15;
    const int nl = jt * 16 + col;  // 0..511
    // mem_in column 256 (tension) rank-1 term + biases
    const float w256r = Wih[(size_t)nl * 257 + 256];
    const float w256z = Wih[(size_t)(512 + nl) * 257 + 256];
    const float w256n = Wih[(size_t)(1024 + nl) * 257 + 256];
    const float brz = bih[nl] + bhh[nl];
    const float bzz = bih[512 + nl] + bhh[512 + nl];
    const float bin = bih[1024 + nl];
    const float bhn = bhh[1024 + nl];
    const int mrow = (lane >> 4) * 8;
#pragma unroll
    for (int v = 0; v < 8; ++v) {
#pragma unroll
      for (int mh2 = 0; mh2 < 2; ++mh2) {
        int m = mh2 * 16 + mrow + v;
        float tm = tensL[m];
        float hold = hiddens[(size_t)(row0 + m) * HID_ + nl];
        float rr = (mh2 ? r1[v] : r0[v]) + tm * w256r + brz;
        float zz = (mh2 ? z1[v] : z0[v]) + tm * w256z + bzz;
        float ii = (mh2 ? i1[v] : i0[v]) + tm * w256n + bin;
        float hh = (mh2 ? hn1[v] : hn0[v]) + bhn;
        rr = 1.f / (1.f + expf(-rr));
        zz = 1.f / (1.f + expf(-zz));
        float nn = tanhf(ii + rr * hh);
        float nh = (1.f - zz) * nn + zz * hold;
        nh *= boostL[m];
        nh = fminf(10.f, fmaxf(-10.f, nh));
        newh[(size_t)(row0 + m) * HID_ + nl] = nh;
      }
    }
  }
}

// ---------------------------------------------------------------------------
// Quadrant segment sums / means / blend (+ family sums), family blend
// ---------------------------------------------------------------------------
__global__ __launch_bounds__(256) void k_qsum(
    const float* __restrict__ newh, const int* __restrict__ npos,
    float* __restrict__ qsum, float* __restrict__ qcnt) {
  __shared__ float qs[16 * 512];
  __shared__ float qc[16];
  int tid = threadIdx.x;
  for (int i = tid; i < 16 * 512; i += 256) qs[i] = 0.f;
  if (tid < 16) qc[tid] = 0.f;
  __syncthreads();
  int agent = blockIdx.x * 256 + tid;
  int qid = (npos[2 * agent] >> 7) * 4 + (npos[2 * agent + 1] >> 7);
  atomicAdd(&qc[qid], 1.f);
  for (int c = 0; c < 512; ++c)
    atomicAdd(&qs[qid * 512 + c], newh[(size_t)agent * 512 + c]);
  __syncthreads();
  for (int i = tid; i < 16 * 512; i += 256)
    if (qs[i] != 0.f) atomicAdd(&qsum[i], qs[i]);
  if (tid < 16) atomicAdd(&qcnt[tid], qc[tid]);
}

__global__ void k_qmean(const float* qsum, const float* qcnt, float* qmean) {
  int i = blockIdx.x * 256 + threadIdx.x;
  if (i < 16 * 512) qmean[i] = qsum[i] / fmaxf(qcnt[i >> 9], 1.f);
}

__global__ __launch_bounds__(256) void k_qapply(
    float* __restrict__ newh, const int* __restrict__ npos,
    const float* __restrict__ qmean, const float* __restrict__ qcnt,
    const float* __restrict__ red, float* __restrict__ famsum) {
  __shared__ int   qidL[256];
  __shared__ float okL[256];
  int tid = threadIdx.x;
  int base = blockIdx.x * 256;
  {
    int agent = base + tid;
    int q = (npos[2 * agent] >> 7) * 4 + (npos[2 * agent + 1] >> 7);
    qidL[tid] = q;
    okL[tid] = (qcnt[q] >= 2.f) ? 1.f : 0.f;
  }
  __syncthreads();
  float sync = 0.1f + 0.1f * (1.f - red[2] / ((float)N_AG * PI_F));
  int c0 = tid * 2, c1 = tid * 2 + 1;
  float f0 = 0.f, f1 = 0.f;
  for (int a = 0; a < 256; ++a) {
    size_t off = (size_t)(base + a) * 512;
    int q = qidL[a];
    float v0 = newh[off + c0], v1 = newh[off + c1];
    if (okL[a] > 0.f) {
      v0 = (1.f - sync) * v0 + sync * qmean[q * 512 + c0];
      v1 = (1.f - sync) * v1 + sync * qmean[q * 512 + c1];
    }
    newh[off + c0] = v0; newh[off + c1] = v1;
    f0 += v0; f1 += v1;
  }
  int fam = base >> 13;  // 256-agent blocks never straddle 8192-agent families
  atomicAdd(&famsum[fam * 512 + c0], f0);
  atomicAdd(&famsum[fam * 512 + c1], f1);
}

__global__ void k_go(const float* famsum, float* go) {
  int c = blockIdx.x * 256 + threadIdx.x;
  if (c < 512) {
    float s = 0.f;
    for (int f = 0; f < 8; ++f) s += famsum[f * 512 + c];
    go[c] = s * (1.f / (float)N_AG);
  }
}

__global__ void k_fapply(float* __restrict__ newh, const float* __restrict__ famsum,
                         const float* __restrict__ go, const int* __restrict__ step) {
  const size_t total = (size_t)N_AG * HID_;
  int st = step[0];
  for (size_t i = (size_t)blockIdx.x * 256 + threadIdx.x; i < total;
       i += (size_t)gridDim.x * 256) {
    size_t agent = i >> 9; int c = (int)(i & 511);
    int fam = (int)(agent >> 13);
    float fm = famsum[fam * 512 + c] * (1.f / 8192.f);
    float v = 0.85f * newh[i] + 0.15f * fm;
    if (((agent & 8191) < 2048) && (st > 5)) v = 0.85f * v + 0.15f * go[c];
    newh[i] = v;
  }
}

// ---------------------------------------------------------------------------
// Softmax over tension, weighted combine, final matvec
// ---------------------------------------------------------------------------
__global__ __launch_bounds__(256) void k_s1(const float* __restrict__ tens, float* red) {
  __shared__ float sb[256];
  int i = blockIdx.x * 256 + threadIdx.x;
  float t = tens[i];
  sb[threadIdx.x] = t; __syncthreads();
  for (int s = 128; s > 0; s >>= 1) {
    if (threadIdx.x < s) sb[threadIdx.x] = fmaxf(sb[threadIdx.x], sb[threadIdx.x + s]);
    __syncthreads();
  }
  if (threadIdx.x == 0) atomicMax((unsigned int*)&red[4], __float_as_uint(sb[0]));
  __syncthreads();
  float s2 = block_reduce_sum(t, sb);
  if (threadIdx.x == 0) atomicAdd(&red[6], s2);
}

__global__ __launch_bounds__(256) void k_s2(const float* __restrict__ tens,
                                            const float* __restrict__ out_ws,
                                            float* red, float* __restrict__ comb) {
  __shared__ float wL[256];
  int tid = threadIdx.x;
  int base = blockIdx.x * 256;
  float tmax = __uint_as_float(((const unsigned int*)red)[4]);
  wL[tid] = expf(tens[base + tid] - tmax);
  __syncthreads();
  float acc = 0.f;
  for (int a = 0; a < 256; ++a)
    acc += wL[a] * out_ws[(size_t)(base + a) * OUT_ + tid];
  atomicAdd(&comb[tid], acc);
  if (tid == 0) {
    float s = 0.f;
    for (int a = 0; a < 256; ++a) s += wL[a];
    atomicAdd(&red[5], s);
  }
}

__global__ void k_pred(const float* __restrict__ comb, const float* __restrict__ red,
                       const float* __restrict__ Woh, const float* __restrict__ boh,
                       float* __restrict__ pred) {
  __shared__ float cn[256];
  int o = threadIdx.x;
  cn[o] = comb[o] / red[5];
  __syncthreads();
  float acc = boh[o];
  for (int c = 0; c < 256; ++c) acc += cn[c] * Woh[o * 256 + c];
  pred[o] = acc;
  if (o == 0) pred[256] = red[6] / (float)N_AG;  // avg_t
}

// ---------------------------------------------------------------------------
extern "C" void kernel_launch(void* const* d_in, const int* in_sizes, int n_in,
                              void* d_out, int out_size, void* d_ws, size_t ws_size,
                              hipStream_t stream) {
  const float* x        = (const float*)d_in[0];
  const float* hiddens  = (const float*)d_in[1];
  const float* grid_in  = (const float*)d_in[2];
  const int*   pos      = (const int*)d_in[3];
  const float* headings = (const float*)d_in[4];
  const int*   ptype    = (const int*)d_in[5];
  const float* noise    = (const float*)d_in[6];
  const int*   step     = (const int*)d_in[7];
  const float* Wa1 = (const float*)d_in[8];
  const float* ba1 = (const float*)d_in[9];
  const float* Wa2 = (const float*)d_in[10];
  const float* ba2 = (const float*)d_in[11];
  const float* Wg1 = (const float*)d_in[12];
  const float* bg1 = (const float*)d_in[13];
  const float* Wg2 = (const float*)d_in[14];
  const float* bg2 = (const float*)d_in[15];
  const float* Wih = (const float*)d_in[16];
  const float* Whh = (const float*)d_in[17];
  const float* bih = (const float*)d_in[18];
  const float* bhh = (const float*)d_in[19];
  const float* Woh = (const float*)d_in[20];
  const float* boh = (const float*)d_in[21];

  char* ws = (char*)d_ws;
  _Float16* wa1p = (_Float16*)(ws + O_WA1P);
  _Float16* wg1p = (_Float16*)(ws + O_WG1P);
  _Float16* wa2p = (_Float16*)(ws + O_WA2P);
  _Float16* wg2p = (_Float16*)(ws + O_WG2P);
  _Float16* wihp = (_Float16*)(ws + O_WIHP);
  _Float16* whhp = (_Float16*)(ws + O_WHHP);
  float* tensw  = (float*)(ws + O_TENS);
  float* headw  = (float*)(ws + O_HEAD);
  int*   nposw  = (int*)(ws + O_NPOS);
  float* gridw  = (float*)(ws + O_GRIDW);
  float* qmean  = (float*)(ws + O_QMEAN);
  float* go     = (float*)(ws + O_GO);
  float* qsum   = (float*)(ws + O_QSUM);
  float* qcnt   = (float*)(ws + O_QCNT);
  float* fam    = (float*)(ws + O_FAM);
  float* red    = (float*)(ws + O_RED);
  float* comb   = (float*)(ws + O_COMB);
  float* out_ws = (float*)(ws + O_OUTWS);

  float* dout  = (float*)d_out;
  float* pred  = dout;                              // 256
  float* newh  = dout + 257;                        // N*HID
  float* goutg = dout + 257 + (size_t)N_AG * HID_;  // NP*GS*GS

  // zero accumulators (every call; harness does not re-poison)
  k_zero<<<(int)((ZERO_FLOATS + 255) / 256), 256, 0, stream>>>((float*)(ws + O_ZERO),
                                                               (int)ZERO_FLOATS);
  // repack weights to f16 WMMA B-fragments
  k_repack<<<(8 * 24 * 32 + 255) / 256, 256, 0, stream>>>(Wa1, wa1p, 8, 24, 768);
  k_repack<<<(8 * 24 * 32 + 255) / 256, 256, 0, stream>>>(Wg1, wg1p, 8, 24, 768);
  k_repack<<<(16 * 4 * 32 + 255) / 256, 256, 0, stream>>>(Wa2, wa2p, 16, 4, 128);
  k_repack<<<(16 * 4 * 32 + 255) / 256, 256, 0, stream>>>(Wg2, wg2p, 16, 4, 128);
  k_repack<<<(96 * 8 * 32 + 255) / 256, 256, 0, stream>>>(Wih, wihp, 96, 8, 257);
  k_repack<<<(96 * 16 * 32 + 255) / 256, 256, 0, stream>>>(Whh, whhp, 96, 16, 512);
  // stigmergy
  k_gridcopy<<<NP_ * GS * GS / 256, 256, 0, stream>>>(grid_in, gridw);
  k_agents<<<N_AG / 256, 256, 0, stream>>>(grid_in, pos, headings, noise, headw, nposw, red);
  k_mh<<<1, 1, 0, stream>>>(red);
  k_coord<<<N_AG / 256, 256, 0, stream>>>(headw, red);
  k_deposit<<<N_AG / 256, 256, 0, stream>>>(ptype, nposw, gridw);
  k_diffuse<<<NP_ * GS * GS / 256, 256, 0, stream>>>(gridw, goutg);
  // fused MLP + GRU (WMMA)
  k_core<<<N_AG / 32, 256, 0, stream>>>(x, hiddens, Wih, ba1, bg1, ba2, bg2, bih, bhh,
                                        wa1p, wg1p, wa2p, wg2p, wihp, whhp,
                                        nposw, goutg, out_ws, tensw, newh);
  // quadrant + family blending
  k_qsum<<<N_AG / 256, 256, 0, stream>>>(newh, nposw, qsum, qcnt);
  k_qmean<<<32, 256, 0, stream>>>(qsum, qcnt, qmean);
  k_qapply<<<N_AG / 256, 256, 0, stream>>>(newh, nposw, qmean, qcnt, red, fam);
  k_go<<<2, 256, 0, stream>>>(fam, go);
  k_fapply<<<8192, 256, 0, stream>>>(newh, fam, go, step);
  // softmax-weighted prediction
  k_s1<<<N_AG / 256, 256, 0, stream>>>(tensw, red);
  k_s2<<<N_AG / 256, 256, 0, stream>>>(tensw, out_ws, red, comb);
  k_pred<<<1, 256, 0, stream>>>(comb, red, Woh, boh, pred);
}
